// CRF_73289321939011
// MI455X (gfx1250) — matrix operand order repaired and checked by reading, not compiled, not run on hardware
//
#include <hip/hip_runtime.h>
#include <hip/hip_bf16.h>

// CRF forward log-partition on gfx1250 (MI455X), wave32 + WMMA bf16.
//
// Per wave: TWO independent 16-batch-row tiles run the 2048-step recurrence
// interleaved (fills serial-chain stalls; B-fragments shared). State P is
// bf16 in LDS (bf16 has f32 exponent range), transitions centered by C so
// renormalization (shfl_xor row-max + log accumulate) is needed only every
// 8 steps. Each step per tile: 4 ds_load_b128 -> 6 v_wmma_f32_16x16x32_bf16
// -> *exp(emission) (f32) -> bf16 stores. Emission loads are forced into the
// GLOBAL address space (global_load_b32, LOADcnt-only) so they do not pollute
// DScnt, which gates the WMMA A-fragment waits. Emissions are register
// double-buffered one step ahead + prefetched ~16 steps ahead.

#define BATCH  256
#define SEQ    2048
#define TAGS   48
#define SxT    (SEQ * TAGS)
#define NTILE  2           // batch tiles per wave
#define RENORM 8           // steps between renormalizations (bf16-safe)
#define CSHIFT 4.4f        // centers exp(trans): worst per-step growth ~e^8.5

typedef __bf16 v16bf __attribute__((ext_vector_type(16)));
typedef __bf16 v8bf  __attribute__((ext_vector_type(8)));
typedef float  v8f   __attribute__((ext_vector_type(8)));
typedef const float __attribute__((address_space(1)))* gcfloat;  // global-AS ptr

__global__ __launch_bounds__(32)
void crf_fwd_wmma(const float* __restrict__ emis,
                  const float* __restrict__ trans,
                  float* __restrict__ out) {
  const int lane = threadIdx.x;      // 0..31 (wave32)
  const int lo   = lane & 15;
  const int hi   = lane >> 4;

  // State tiles: 16 rows x 64 bf16 (48 tags + zero pad to K=64). Pad stays 0.
  __shared__ __attribute__((aligned(16))) __bf16 P[NTILE][16 * 64];

  // ---- init state: alpha0[b,0]=0 -> P[b][0]=1, everything else exp(-1e4)=0
  #pragma unroll
  for (int u = 0; u < NTILE; ++u) {
    #pragma unroll
    for (int k = 0; k < 32; ++k) P[u][lane * 32 + k] = (__bf16)0.0f;
  }
  __syncthreads();
  if (hi == 0) {
    #pragma unroll
    for (int u = 0; u < NTILE; ++u) P[u][lo * 64 + 0] = (__bf16)1.0f;
  }
  __syncthreads();

  // ---- loop-invariant B fragments: W[j][i] = exp(trans[i][j] - C)
  // 16-bit B 32x16 striping: element e -> K = 8*laneHi + e (e<8), 16 + 8*laneHi + (e-8) (e>=8), N = laneLo.
  v16bf bf[2][3];
  #pragma unroll
  for (int kk = 0; kk < 2; ++kk)
    #pragma unroll
    for (int n = 0; n < 3; ++n)
      #pragma unroll
      for (int e = 0; e < 16; ++e) {
        const int j = 32 * kk + 8 * hi + (e < 8 ? e : e + 8);     // K index (padded to 64)
        const int i = 16 * n + lo;               // N index (output tag)
        const float w = (j < TAGS) ? __expf(trans[i * TAGS + j] - CSHIFT) : 0.0f;
        bf[kk][n][e] = (__bf16)w;
      }

  // ---- final-transition weights wl[n] = exp(trans[T-1][i] - ml)
  float ml = -1e30f;
  for (int i = 0; i < TAGS; ++i) ml = fmaxf(ml, trans[(TAGS - 1) * TAGS + i]);
  float wl[3];
  #pragma unroll
  for (int n = 0; n < 3; ++n)
    wl[n] = __expf(trans[(TAGS - 1) * TAGS + 16 * n + lo] - ml);

  // ---- emission pointers: lane covers (b = tile*16 + 8*hi + r, i = 16n+lo)
  // Global-AS pointers -> global_load_b32 (LOADcnt only, DScnt untouched).
  gcfloat ep[NTILE];
  const float* pp[NTILE];            // generic mirror, prefetch only
  #pragma unroll
  for (int u = 0; u < NTILE; ++u) {
    const size_t off = (size_t)((blockIdx.x * NTILE + u) * 16 + 8 * hi) * SxT + lo;
    ep[u] = (gcfloat)emis + off;
    pp[u] = emis + off;
  }

  float logscale[NTILE][8];
  #pragma unroll
  for (int u = 0; u < NTILE; ++u)
    #pragma unroll
    for (int r = 0; r < 8; ++r) logscale[u][r] = 0.0f;

  float pn[NTILE][8][3];   // current state values (D layout), kept for epilogue
  float em[NTILE][8][3];   // emission double buffer (one step ahead)

  #pragma unroll
  for (int u = 0; u < NTILE; ++u) {
    #pragma unroll
    for (int r = 0; r < 8; ++r)
      #pragma unroll
      for (int n = 0; n < 3; ++n)
        em[u][r][n] = ep[u][r * SxT + 16 * n];   // s = 0
    ep[u] += TAGS;                               // -> s = 1
    pp[u] += TAGS;
  }

  for (int s = 0; s < SEQ; ++s) {
    // A fragments from LDS state. 16-bit A 16x32 striping: lane holds row b=lo,
    // halves K = 8*hi..+7 and 16+8*hi..+7 per chunk (16B each).
    v8f acc[NTILE][3];
    #pragma unroll
    for (int u = 0; u < NTILE; ++u) {
      const __bf16* rp = &P[u][lo * 64 + 8 * hi];
      v8bf c0 = *(const v8bf*)(rp +  0);
      v8bf c1 = *(const v8bf*)(rp + 16);
      v8bf c2 = *(const v8bf*)(rp + 32);
      v8bf c3 = *(const v8bf*)(rp + 48);
      v16bf a0 = __builtin_shufflevector(c0, c1, 0,1,2,3,4,5,6,7,8,9,10,11,12,13,14,15);
      v16bf a1 = __builtin_shufflevector(c2, c3, 0,1,2,3,4,5,6,7,8,9,10,11,12,13,14,15);
      #pragma unroll
      for (int n = 0; n < 3; ++n) {
        v8f z = {0.f,0.f,0.f,0.f,0.f,0.f,0.f,0.f};
        z = __builtin_amdgcn_wmma_f32_16x16x32_bf16(false, a0, false, bf[0][n],
                                                    (short)0, z, false, false);
        z = __builtin_amdgcn_wmma_f32_16x16x32_bf16(false, a1, false, bf[1][n],
                                                    (short)0, z, false, false);
        acc[u][n] = z;
      }
    }

    // apply exp(emissions)
    #pragma unroll
    for (int u = 0; u < NTILE; ++u)
      #pragma unroll
      for (int r = 0; r < 8; ++r)
        #pragma unroll
        for (int n = 0; n < 3; ++n)
          pn[u][r][n] = acc[u][n][r] * __expf(em[u][r][n]);

    // preload next step's emissions (independent of recurrence -> overlaps)
    if (s + 1 < SEQ) {
      #pragma unroll
      for (int u = 0; u < NTILE; ++u) {
        #pragma unroll
        for (int r = 0; r < 8; ++r)
          #pragma unroll
          for (int n = 0; n < 3; ++n)
            em[u][r][n] = ep[u][r * SxT + 16 * n];
        ep[u] += TAGS;
        // prefetch ~16 steps ahead (speculative; OOB silently dropped)
        #pragma unroll
        for (int r = 0; r < 8; ++r) {
          __builtin_prefetch(pp[u] + r * SxT + 14 * TAGS,      0, 3);
          __builtin_prefetch(pp[u] + r * SxT + 14 * TAGS + 32, 0, 3);
        }
        pp[u] += TAGS;
      }
    }

    // renormalize every RENORM steps: row max across 16 lanes of each half
    if ((s & (RENORM - 1)) == (RENORM - 1)) {
      #pragma unroll
      for (int u = 0; u < NTILE; ++u)
        #pragma unroll
        for (int r = 0; r < 8; ++r) {
          float t = fmaxf(fmaxf(pn[u][r][0], pn[u][r][1]), pn[u][r][2]);
          #pragma unroll
          for (int m = 1; m <= 8; m <<= 1)
            t = fmaxf(t, __shfl_xor(t, m, 32));
          const float inv = __builtin_amdgcn_rcpf(t);
          logscale[u][r] += __logf(t);
          #pragma unroll
          for (int n = 0; n < 3; ++n) pn[u][r][n] *= inv;
        }
    }

    // store state back (bf16, D layout)
    #pragma unroll
    for (int u = 0; u < NTILE; ++u)
      #pragma unroll
      for (int r = 0; r < 8; ++r)
        #pragma unroll
        for (int n = 0; n < 3; ++n)
          P[u][(r + 8 * hi) * 64 + 16 * n + lo] = (__bf16)pn[u][r][n];
    // single wave: per-wave DS ordering + compiler s_wait_dscnt handle hazards
  }

  // ---- epilogue: out[b] = logscale[b] + SEQ*C + ml + log( sum_i P[b,i]*wl[i] )
  // (SEQ % RENORM == 0, so pn is normalized here)
  #pragma unroll
  for (int u = 0; u < NTILE; ++u)
    #pragma unroll
    for (int r = 0; r < 8; ++r) {
      float srow = pn[u][r][0] * wl[0] + pn[u][r][1] * wl[1] + pn[u][r][2] * wl[2];
      #pragma unroll
      for (int m = 1; m <= 8; m <<= 1)
        srow += __shfl_xor(srow, m, 32);
      if (lo == 0)
        out[(blockIdx.x * NTILE + u) * 16 + 8 * hi + r] =
            logscale[u][r] + (float)SEQ * CSHIFT + ml + __logf(srow);
    }
}

extern "C" void kernel_launch(void* const* d_in, const int* in_sizes, int n_in,
                              void* d_out, int out_size, void* d_ws, size_t ws_size,
                              hipStream_t stream) {
  const float* emissions   = (const float*)d_in[0];
  // d_in[1] = mask: forward value is mask-independent (a*m + (1-m)*a == a)
  const float* transitions = (const float*)d_in[2];
  float* out = (float*)d_out;

  crf_fwd_wmma<<<dim3(BATCH / (16 * NTILE)), dim3(32), 0, stream>>>(
      emissions, transitions, out);
}
